// SparseAutoencoderTopK_67284957659716
// MI455X (gfx1250) — matrix operand order repaired and checked by reading, not compiled, not run on hardware
//
#include <hip/hip_runtime.h>
#include <hip/hip_bf16.h>
#include <stdint.h>

// ---------------------------------------------------------------------------
// Problem constants: B=8192, D_IN=768, D_LAT=12288, K=64
// ---------------------------------------------------------------------------
#define B_ROWS 8192
#define D_IN   768
#define D_LAT  12288
#define K_TOP  64

// Encoder GEMM tiling
#define BM 128
#define BN 256
#define BK 32
#define NK (D_IN / BK)   // 24
#define LDT 40           // LDS tile pitch in bf16 (80B = 16B-aligned; 20*r mod 64 is a bank permutation)

typedef __attribute__((ext_vector_type(16))) __bf16        v16bf;
typedef __attribute__((ext_vector_type(8)))  float         v8f;
typedef __attribute__((ext_vector_type(4)))  unsigned int  v4u;
typedef __attribute__((ext_vector_type(8)))  int           v8i;
typedef __attribute__((ext_vector_type(4)))  int           v4i;

#if defined(__has_builtin)
#if __has_builtin(__builtin_amdgcn_tensor_load_to_lds) && __has_builtin(__builtin_amdgcn_s_wait_tensorcnt)
#define HAVE_TDM 1
#endif
#endif

__device__ __forceinline__ unsigned short f32_to_bf16_rne(float f) {
    unsigned int u = __float_as_uint(f);
    u += 0x7FFFu + ((u >> 16) & 1u);
    return (unsigned short)(u >> 16);
}

__device__ __forceinline__ unsigned int lds_offset_of(const void* p) {
    // generic pointers to LDS carry the LDS byte offset in their low 32 bits
    return (unsigned int)(uintptr_t)p;
}

#ifdef HAVE_TDM
// Issue one TDM 2D tile load: tile_d1 rows x tile_d0 bf16 elements, row stride
// tensor_d0 elements in global; LDS rows padded 64B->80B (pad 4 dwords per 16).
// D# layout per CDNA5 ISA.
__device__ __forceinline__ void tdm_load_tile_2d(unsigned int lds_byte_off,
                                                 const unsigned short* gsrc,
                                                 unsigned int tensor_d0,
                                                 unsigned int tensor_d1,
                                                 unsigned int tile_d0,
                                                 unsigned int tile_d1) {
    unsigned long long ga = (unsigned long long)(uintptr_t)gsrc;
    v4u g0;
    g0.x = 1u;                                             // count=1 (valid), user mode
    g0.y = lds_byte_off;                                   // lds_addr [63:32]
    g0.z = (unsigned int)(ga & 0xFFFFFFFFu);               // global_addr [95:64]
    g0.w = (unsigned int)((ga >> 32) & 0x01FFFFFFu)        // global_addr [120:96]
         | (2u << 30);                                     // type=2 ("image") [127:126]

    const unsigned int s0 = (1u << 16)                     // data_size=1 -> 2 bytes
                          | (1u << 20)                     // pad_enable
                          | (3u << 22)                     // pad_interval: 16 dwords (64B)
                          | (3u << 25);                    // pad_amount: 4 dwords (16B)
    const unsigned int s1 = (tensor_d0 & 0xFFFFu) << 16;                       // tensor_dim0[15:0]
    const unsigned int s2 = (tensor_d0 >> 16) | ((tensor_d1 & 0xFFFFu) << 16); // dim0 hi | dim1 lo
    const unsigned int s3 = (tensor_d1 >> 16) | (tile_d0 << 16);               // dim1 hi | tile_dim0
    const unsigned int s4 = tile_d1 & 0xFFFFu;                                 // tile_dim1 (tile_dim2=0)
    const unsigned int s5 = tensor_d0;                                         // tensor_dim0_stride lo32
    v8i g1;
    g1[0] = (int)s0; g1[1] = (int)s1; g1[2] = (int)s2; g1[3] = (int)s3;
    g1[4] = (int)s4; g1[5] = (int)s5; g1[6] = 0; g1[7] = 0;
    v4i g2 = {0, 0, 0, 0};                                  // 2D tile: dims 2..4 unused
    v4i g3 = {0, 0, 0, 0};
#if __clang_major__ >= 23
    v8i g4 = {0, 0, 0, 0, 0, 0, 0, 0};
    __builtin_amdgcn_tensor_load_to_lds(g0, g1, g2, g3, g4, 0);
#else
    __builtin_amdgcn_tensor_load_to_lds(g0, g1, g2, g3, 0);
#endif
}
#endif // HAVE_TDM

// ---------------------------------------------------------------------------
// Kernel 0a: f32 -> bf16 (x)
// ---------------------------------------------------------------------------
__global__ __launch_bounds__(256) void cvt_bf16_kernel(const float* __restrict__ in,
                                                       unsigned short* __restrict__ out,
                                                       int n) {
    int i = blockIdx.x * 256 + threadIdx.x;
    if (i < n) out[i] = f32_to_bf16_rne(in[i]);
}

// ---------------------------------------------------------------------------
// Kernel 0b: W_enc [768][12288] f32 -> wbT [12288][768] bf16 (tiled transpose)
// grid (12288/32, 768/32), block 256
// ---------------------------------------------------------------------------
__global__ __launch_bounds__(256) void cvt_transpose_kernel(const float* __restrict__ in,
                                                            unsigned short* __restrict__ out) {
    __shared__ unsigned short t[32][33];
    const int n0 = blockIdx.x * 32;
    const int k0 = blockIdx.y * 32;
    const int c = threadIdx.x & 31;
    const int r4 = (threadIdx.x >> 5) * 4;
#pragma unroll
    for (int j = 0; j < 4; ++j) {
        const int r = r4 + j;
        t[r][c] = f32_to_bf16_rne(in[(size_t)(k0 + r) * D_LAT + n0 + c]);
    }
    __syncthreads();
#pragma unroll
    for (int j = 0; j < 4; ++j) {
        const int r = r4 + j;
        out[(size_t)(n0 + r) * D_IN + k0 + c] = t[c][r];
    }
}

// ---------------------------------------------------------------------------
// Kernel 1: encoder GEMM  h = x @ W_enc + b_enc
//   grid (D_LAT/BN, B_ROWS/BM) = (48, 64), block 256 = 8 wave32 waves (2x4)
//   per wave: 64(M) x 64(N) = 4x4 v_wmma_f32_16x16x32_bf16 accumulators
//   A/B tiles double-buffered in LDS; staged by the Tensor Data Mover
//   (wave 0 issues tensor_load_to_lds, s_wait_tensorcnt + barrier to publish)
// ---------------------------------------------------------------------------
__global__ __launch_bounds__(256) void enc_gemm_kernel(const unsigned short* __restrict__ xb,
                                                       const unsigned short* __restrict__ wbT,
                                                       const float* __restrict__ b_enc,
                                                       float* __restrict__ h) {
    __shared__ unsigned short As[2][BM * LDT];   // [row][k], 80B pitch
    __shared__ unsigned short Bs[2][BN * LDT];   // [col][k], 80B pitch

    const int tid   = threadIdx.x;
    const int lane  = tid & 31;
    const int wid   = tid >> 5;
    const int waveM = wid >> 2;                  // 0..1
    const int waveN = wid & 3;                   // 0..3
    const int l16   = lane & 15;
    const int khalf = lane >> 4;                 // 0/1
    const int n0    = blockIdx.x * BN;
    const int m0    = blockIdx.y * BM;

    union Frag { v16bf v; unsigned int u[8]; };
    union Acc  { v8f  v; float f[8]; };

    Acc acc[4][4];
#pragma unroll
    for (int mi = 0; mi < 4; ++mi)
#pragma unroll
        for (int ni = 0; ni < 4; ++ni)
#pragma unroll
            for (int q = 0; q < 8; ++q) acc[mi][ni].f[q] = 0.0f;

    // ---- tile staging ----
    auto stage = [&](int buf, int k0) {
#ifdef HAVE_TDM
        if (wid == 0) {
            tdm_load_tile_2d(lds_offset_of(&As[buf][0]), xb  + (size_t)m0 * D_IN + k0,
                             D_IN, B_ROWS, BK, BM);
            tdm_load_tile_2d(lds_offset_of(&Bs[buf][0]), wbT + (size_t)n0 * D_IN + k0,
                             D_IN, D_LAT, BK, BN);
        }
#else
        // fallback: 256 threads copy 16B chunks (both tiles are K-contiguous)
#pragma unroll
        for (int i = 0; i < 2; ++i) {            // A: 512 chunks
            const int chunk = tid + 256 * i;
            const int row = chunk >> 2;
            const int c8  = (chunk & 3) * 8;
            *(uint4*)&As[buf][row * LDT + c8] =
                *(const uint4*)(xb + (size_t)(m0 + row) * D_IN + k0 + c8);
        }
#pragma unroll
        for (int i = 0; i < 4; ++i) {            // B: 1024 chunks
            const int chunk = tid + 256 * i;
            const int col = chunk >> 2;
            const int c8  = (chunk & 3) * 8;
            *(uint4*)&Bs[buf][col * LDT + c8] =
                *(const uint4*)(wbT + (size_t)(n0 + col) * D_IN + k0 + c8);
        }
#endif
    };
    auto publish = [&]() {                       // make staged tiles visible to all waves
#ifdef HAVE_TDM
        if (wid == 0) __builtin_amdgcn_s_wait_tensorcnt(0);
#endif
        __syncthreads();
    };

    stage(0, 0);
    publish();

    for (int k = 0; k < NK; ++k) {
        const int cur = k & 1;
        if (k + 1 < NK) stage(cur ^ 1, (k + 1) * BK);

        // B fragments: col N = lane&15; u[0..3] = K[khalf*8 + 0..7], u[4..7] = K[16 + khalf*8 + 0..7]
        Frag bfr[4];
#pragma unroll
        for (int ni = 0; ni < 4; ++ni) {
            const int col = waveN * 64 + ni * 16 + l16;
            const unsigned int* p = (const unsigned int*)&Bs[cur][col * LDT + khalf * 8];
            const unsigned int* p1 = (const unsigned int*)&Bs[cur][col * LDT + 16 + khalf * 8];
#pragma unroll
            for (int q = 0; q < 4; ++q) { bfr[ni].u[q] = p[q]; bfr[ni].u[4 + q] = p1[q]; }
        }
        // A fragments: u[0..3] = K[khalf*8 + 0..7], u[4..7] = K[16 + khalf*8 + 0..7]
#pragma unroll
        for (int mi = 0; mi < 4; ++mi) {
            const int row = waveM * 64 + mi * 16 + l16;
            Frag afr;
            const unsigned int* pa0 = (const unsigned int*)&As[cur][row * LDT + khalf * 8];
            const unsigned int* pa1 = (const unsigned int*)&As[cur][row * LDT + 16 + khalf * 8];
#pragma unroll
            for (int q = 0; q < 4; ++q) { afr.u[q] = pa0[q]; afr.u[4 + q] = pa1[q]; }
#pragma unroll
            for (int ni = 0; ni < 4; ++ni) {
                acc[mi][ni].v = __builtin_amdgcn_wmma_f32_16x16x32_bf16(
                    false, afr.v, false, bfr[ni].v, (short)0, acc[mi][ni].v, false, false);
            }
        }
        if (k + 1 < NK) publish();
    }

    // ---- epilogue: + b_enc, write h (C layout: M = v + khalf*8, N = lane&15) ----
#pragma unroll
    for (int ni = 0; ni < 4; ++ni) {
        const int n = n0 + waveN * 64 + ni * 16 + l16;
        const float bias = b_enc[n];
#pragma unroll
        for (int mi = 0; mi < 4; ++mi) {
            const int rbase = m0 + waveM * 64 + mi * 16 + khalf * 8;
#pragma unroll
            for (int vv = 0; vv < 8; ++vv)
                h[(size_t)(rbase + vv) * D_LAT + n] = acc[mi][ni].f[vv] + bias;
        }
    }
}

// ---------------------------------------------------------------------------
// Kernel 2: exact top-64 per row (4-pass byte radix-select, deterministic)
// ---------------------------------------------------------------------------
__global__ __launch_bounds__(256) void topk_kernel(float* __restrict__ z,
                                                   float* __restrict__ nzval,
                                                   int* __restrict__ nzidx) {
    const int row = blockIdx.x;
    const int tid = threadIdx.x;
    float* zr = z + (size_t)row * D_LAT;
    constexpr int NPT = D_LAT / 256;             // 48

    float v[NPT];
    unsigned int key[NPT];
#pragma unroll
    for (int i = 0; i < NPT; ++i) {
        const float f = zr[tid + i * 256];
        v[i] = f;
        const unsigned int u = __float_as_uint(f);
        key[i] = u ^ ((u >> 31) ? 0xFFFFFFFFu : 0x80000000u);
    }

    __shared__ int hist[256];
    __shared__ int scan[256];
    __shared__ int sh_bin, sh_cum;

    unsigned int prefix = 0;
    int need = K_TOP;
    for (int shift = 24; shift >= 0; shift -= 8) {
        hist[tid] = 0;
        __syncthreads();
#pragma unroll
        for (int i = 0; i < NPT; ++i) {
            const bool match = (shift == 24) || (((key[i] ^ prefix) >> (shift + 8)) == 0);
            if (match) atomicAdd(&hist[(key[i] >> shift) & 0xFF], 1);
        }
        __syncthreads();
        if (tid == 0) {
            int cum = 0, b = 255;
            for (; b > 0; --b) {
                if (cum + hist[b] >= need) break;
                cum += hist[b];
            }
            sh_bin = b;
            sh_cum = cum;
        }
        __syncthreads();
        prefix |= ((unsigned int)sh_bin << shift);
        need -= sh_cum;
        __syncthreads();
    }
    const unsigned int T = prefix;

    int tieLocal = 0;
#pragma unroll
    for (int i = 0; i < NPT; ++i) tieLocal += (key[i] == T);
    scan[tid] = tieLocal; __syncthreads();
    for (int off = 1; off < 256; off <<= 1) {
        const int t = (tid >= off) ? scan[tid - off] : 0;
        __syncthreads();
        scan[tid] += t;
        __syncthreads();
    }
    const int tieBefore = scan[tid] - tieLocal;
    __syncthreads();

    int tr = tieBefore, keepLocal = 0;
#pragma unroll
    for (int i = 0; i < NPT; ++i) {
        bool kp = key[i] > T;
        if (key[i] == T) { kp = (tr < need); ++tr; }
        keepLocal += kp;
    }
    scan[tid] = keepLocal; __syncthreads();
    for (int off = 1; off < 256; off <<= 1) {
        const int t = (tid >= off) ? scan[tid - off] : 0;
        __syncthreads();
        scan[tid] += t;
        __syncthreads();
    }
    int slot = scan[tid] - keepLocal;

    tr = tieBefore;
#pragma unroll
    for (int i = 0; i < NPT; ++i) {
        bool kp = key[i] > T;
        if (key[i] == T) { kp = (tr < need); ++tr; }
        const int idx = tid + i * 256;
        zr[idx] = kp ? v[i] : 0.0f;
        if (kp) {
            nzidx[row * K_TOP + slot] = idx;
            nzval[row * K_TOP + slot] = v[i];
            ++slot;
        }
    }
}

// ---------------------------------------------------------------------------
// Kernel 3: sparse decoder  recon = z_sparse @ W_dec + b_dec  (W_dec L2-resident)
// ---------------------------------------------------------------------------
__global__ __launch_bounds__(256) void dec_kernel(const float* __restrict__ nzval,
                                                  const int* __restrict__ nzidx,
                                                  const float* __restrict__ W_dec,
                                                  const float* __restrict__ b_dec,
                                                  float* __restrict__ recon) {
    const int row = blockIdx.x;
    const int tid = threadIdx.x;
    __shared__ float sval[K_TOP];
    __shared__ int   sidx[K_TOP];
    if (tid < K_TOP) {
        sval[tid] = nzval[row * K_TOP + tid];
        sidx[tid] = nzidx[row * K_TOP + tid];
    }
    __syncthreads();

    float a0 = b_dec[tid];
    float a1 = b_dec[tid + 256];
    float a2 = b_dec[tid + 512];
#pragma unroll 4
    for (int j = 0; j < K_TOP; ++j) {
        const float vv = sval[j];
        const float* wr = W_dec + (size_t)sidx[j] * D_IN;
        a0 += vv * wr[tid];
        a1 += vv * wr[tid + 256];
        a2 += vv * wr[tid + 512];
    }
    float* rr = recon + (size_t)row * D_IN;
    rr[tid] = a0; rr[tid + 256] = a1; rr[tid + 512] = a2;
}

// ---------------------------------------------------------------------------
// Host launcher
// ---------------------------------------------------------------------------
extern "C" void kernel_launch(void* const* d_in, const int* in_sizes, int n_in,
                              void* d_out, int out_size, void* d_ws, size_t ws_size,
                              hipStream_t stream) {
    (void)in_sizes; (void)n_in; (void)out_size; (void)ws_size;

    const float* x     = (const float*)d_in[0];
    const float* W_enc = (const float*)d_in[1];
    const float* b_enc = (const float*)d_in[2];
    const float* W_dec = (const float*)d_in[3];
    const float* b_dec = (const float*)d_in[4];
    // d_in[5] = k_top scalar (== 64, compiled in)

    float* recon = (float*)d_out;                                // [8192, 768]
    float* z     = recon + (size_t)B_ROWS * D_IN;                // [8192, 12288]

    unsigned short* xb  = (unsigned short*)d_ws;                 // x bf16      [8192][768]
    unsigned short* wbT = xb + (size_t)B_ROWS * D_IN;            // W_enc^T bf16 [12288][768]
    float* nzval = (float*)(wbT + (size_t)D_LAT * D_IN);         // [8192][64]
    int*   nzidx = (int*)(nzval + (size_t)B_ROWS * K_TOP);       // [8192][64]

    const int nx = B_ROWS * D_IN;
    cvt_bf16_kernel<<<(nx + 255) / 256, 256, 0, stream>>>(x, xb, nx);
    cvt_transpose_kernel<<<dim3(D_LAT / 32, D_IN / 32), 256, 0, stream>>>(W_enc, wbT);

    dim3 ggrid(D_LAT / BN, B_ROWS / BM);                         // (48, 64)
    enc_gemm_kernel<<<ggrid, 256, 0, stream>>>(xb, wbT, b_enc, z);

    topk_kernel<<<B_ROWS, 256, 0, stream>>>(z, nzval, nzidx);

    dec_kernel<<<B_ROWS, 256, 0, stream>>>(nzval, nzidx, W_dec, b_dec, recon);
}
